// TaskOptimizer_13460427505881
// MI455X (gfx1250) — matrix operand order repaired and checked by reading, not compiled, not run on hardware
//
#include <hip/hip_runtime.h>
#include <hip/hip_bf16.h>
#include <stdint.h>

// ---------------- model dims ----------------
#define BB      2
#define SS      1024
#define DD      768
#define HH      12
#define HDD     64
#define MLPD    3072
#define VOCABN  32001
#define CONDN   256
#define LLAYERS 8
#define MROWS   (BB*SS)          // 2048
#define ATT_SCALE 0.125f         // 1/sqrt(64)

typedef __bf16 bf16;
typedef __attribute__((ext_vector_type(16))) __bf16 v16bf;
typedef __attribute__((ext_vector_type(8)))  float  v8f;

#define WMMA_BF16(a,b,c) \
  __builtin_amdgcn_wmma_f32_16x16x32_bf16(false,(a),false,(b),(short)0,(c),false,false)

__device__ __forceinline__ bf16 f2bf(float f){
  union { float f; uint32_t u; } a; a.f = f;
  uint32_t r = a.u + 0x7FFFu + ((a.u >> 16) & 1u);   // round-to-nearest-even
  union { uint16_t u; bf16 b; } o; o.u = (uint16_t)(r >> 16);
  return o.b;
}

// A-fragment (16x32 bf16): lanes 0-15 row M, lanes 0-15 K=0..15, lanes 16-31 K=16..31
__device__ __forceinline__ v16bf load_afrag(const bf16* rowPtr, int hi){
  union { uint4 q[2]; v16bf v; } u;
  u.q[0] = *(const uint4*)(rowPtr + hi*16);
  u.q[1] = *(const uint4*)(rowPtr + hi*16 + 8);
  return u.v;
}
// B-fragment (32x16 bf16): lane holds column N; lanes 0-15 K=0..15, lanes 16-31 K=16..31
__device__ __forceinline__ v16bf load_bfrag(const bf16* colPtr, int hi){
  union { uint4 q[2]; v16bf v; } u;
  u.q[0] = *(const uint4*)(colPtr + hi*16);
  u.q[1] = *(const uint4*)(colPtr + hi*16 + 8);
  return u.v;
}

__device__ __forceinline__ float silu_f(float x){ return x / (1.f + __expf(-x)); }
__device__ __forceinline__ float gelu_tanh(float x){
  float x3 = x*x*x;
  return 0.5f*x*(1.f + tanhf(0.7978845608028654f*(x + 0.044715f*x3)));
}

// ---------------- conditioning: c = silu(silu(temb@W1+b1)@W2+b2) + (silu(score@w1+b1)@w2+b2)
__global__ __launch_bounds__(256) void cond_kernel(
    const float* __restrict__ score, const float* __restrict__ sigma,
    const float* __restrict__ sw1, const float* __restrict__ sb1,
    const float* __restrict__ sw2, const float* __restrict__ sb2,
    const float* __restrict__ cw1, const float* __restrict__ cb1,
    const float* __restrict__ cw2, const float* __restrict__ cb2,
    float* __restrict__ c)
{
  __shared__ float sa[256], sbuf[256];
  const int b = blockIdx.x, t = threadIdx.x;
  float sg = sigma[b];
  int i = t & 127;
  float f = __expf(-0.07195578415606393f * (float)i);   // exp(-ln(1e4)*i/128)
  float ang = sg * f;
  sa[t] = (t < 128) ? cosf(ang) : sinf(ang);
  __syncthreads();
  float acc = sb1[t];
  for (int k = 0; k < 256; k++) acc += sa[k]*sw1[k*256 + t];
  sbuf[t] = silu_f(acc);
  __syncthreads();
  float acc2 = sb2[t];
  for (int k = 0; k < 256; k++) acc2 += sbuf[k]*sw2[k*256 + t];
  float sig_c = silu_f(acc2);
  __syncthreads();
  sa[t] = silu_f(score[b]*cw1[t] + cb1[t]);
  __syncthreads();
  float acc3 = cb2[t];
  for (int k = 0; k < 256; k++) acc3 += sa[k]*cw2[k*256 + t];
  c[b*256 + t] = sig_c + acc3;
}

// ---------------- adaLN mods: out[b, col] = c[b,:] @ W + bias
__global__ __launch_bounds__(256) void ada_kernel(
    const float* __restrict__ c, const float* __restrict__ w,
    const float* __restrict__ bias, float* __restrict__ out, int NO)
{
  __shared__ float cs[256];
  const int b = blockIdx.y;
  cs[threadIdx.x] = c[b*256 + threadIdx.x];
  __syncthreads();
  int col = blockIdx.x*256 + threadIdx.x;
  if (col < NO){
    float acc = bias[col];
    for (int k = 0; k < 256; k++) acc += cs[k]*w[(size_t)k*NO + col];
    out[(size_t)b*NO + col] = acc;
  }
}

// ---------------- embedding gather
__global__ __launch_bounds__(256) void embed_kernel(
    const int* __restrict__ idx, const float* __restrict__ embed, float* __restrict__ x)
{
  const int row = blockIdx.x;
  const int v = idx[row];
  for (int d = threadIdx.x; d < DD; d += 256)
    x[(size_t)row*DD + d] = embed[(size_t)v*DD + d];
}

// ---------------- LayerNorm + adaLN modulation -> h (f32)
__global__ __launch_bounds__(256) void ln_mod_kernel(
    const float* __restrict__ x, const float* __restrict__ wgt,
    const float* __restrict__ mods, int sh_off, int sc_off, int mstride,
    float* __restrict__ h)
{
  __shared__ float rs[256], rq[256];
  const int row = blockIdx.x, tid = threadIdx.x;
  const int b = row / SS;
  const float* xr = x + (size_t)row*DD;
  float vals[3], s0 = 0.f, s1 = 0.f;
  #pragma unroll
  for (int i = 0; i < 3; i++){ float v = xr[tid + i*256]; vals[i] = v; s0 += v; s1 += v*v; }
  rs[tid] = s0; rq[tid] = s1; __syncthreads();
  for (int o = 128; o > 0; o >>= 1){
    if (tid < o){ rs[tid] += rs[tid + o]; rq[tid] += rq[tid + o]; }
    __syncthreads();
  }
  float mean = rs[0] * (1.f/768.f);
  float var  = rq[0] * (1.f/768.f) - mean*mean;
  float inv  = rsqrtf(var + 1e-5f);
  const float* mb = mods + (size_t)b*mstride;
  #pragma unroll
  for (int i = 0; i < 3; i++){
    int d = tid + i*256;
    h[(size_t)row*DD + d] =
      (vals[i]-mean)*inv*wgt[d]*(1.f + mb[sc_off + d]) + mb[sh_off + d];
  }
}

// ---------------- RoPE on q,k inside qkv [M][3*D]
__global__ __launch_bounds__(256) void rope_kernel(float* __restrict__ qkv)
{
  const int row = blockIdx.x;
  const int s = row % SS;
  for (int idx = threadIdx.x; idx < 768; idx += 256){
    int t  = idx / 384;       // 0=q, 1=k
    int p  = idx % 384;
    int hh = p >> 5;
    int d  = p & 31;
    float invf = __expf(-0.28782313662425575f * (float)d);  // 1e4^(-d/32)
    float ang = (float)s * invf;
    float cs = cosf(ang), sn = sinf(ang);
    size_t base = (size_t)row*(3*DD) + (size_t)t*DD + hh*HDD + d;
    float x1 = qkv[base], x2 = qkv[base + 32];
    qkv[base]      = x1*cs - x2*sn;
    qkv[base + 32] = x2*cs + x1*sn;
  }
}

// ---------------- WMMA GEMM: C = epi(A[M,K] @ B[K,N]) ; A,B f32, bf16 compute, f32 accum
// EPI 0: (+bias) store ; EPI 1: gelu(+bias) store ; EPI 2: C += gate[b,col]*( +bias acc)
template<int EPI>
__global__ __launch_bounds__(256) void gemm_wmma(
    const float* __restrict__ A, const float* __restrict__ Bw,
    const float* __restrict__ bias, const float* __restrict__ gate,
    float* __restrict__ C, int N, int K, int gstride)
{
  __shared__ bf16 As[64*40];    // [row][k] pad 40
  __shared__ bf16 Bs[128*40];   // [n][k]  pad 40 (transposed)
  const int tid = threadIdx.x, lane = tid & 31, w = tid >> 5;
  const int hi = lane >> 4, lo = lane & 15;
  const int wm = w & 3, wn = w >> 2;
  const int mb = blockIdx.y * 64, nb = blockIdx.x * 128;
  v8f acc[4] = {};

  const int ar = tid >> 2;            // 0..63
  const int ak = (tid & 3) * 8;       // 0,8,16,24
  const int bk = tid >> 3;            // 0..31
  const int bn = (tid & 7) * 16;      // 0..112

  for (int kk = 0; kk < K; kk += 32){
    // A tile 64x32 -> bf16 LDS
    const float* ap = A + (size_t)(mb + ar)*K + kk + ak;
    float4 a0 = *(const float4*)ap;
    float4 a1 = *(const float4*)(ap + 4);
    if (kk + 32 < K) __builtin_prefetch(ap + 32, 0, 1);
    bf16* asd = &As[ar*40 + ak];
    asd[0]=f2bf(a0.x); asd[1]=f2bf(a0.y); asd[2]=f2bf(a0.z); asd[3]=f2bf(a0.w);
    asd[4]=f2bf(a1.x); asd[5]=f2bf(a1.y); asd[6]=f2bf(a1.z); asd[7]=f2bf(a1.w);
    // B tile 32x128 -> transposed bf16 LDS (guard N edge)
    const float* bp = Bw + (size_t)(kk + bk)*N + nb + bn;
    if (kk + 32 < K) __builtin_prefetch(bp + (size_t)32*N, 0, 1);
    #pragma unroll
    for (int j = 0; j < 16; j++){
      int col = nb + bn + j;
      float v = (col < N) ? bp[j] : 0.f;
      Bs[(bn + j)*40 + bk] = f2bf(v);
    }
    __syncthreads();
    v16bf af = load_afrag(&As[(wm*16 + lo)*40], hi);
    #pragma unroll
    for (int t = 0; t < 4; t++){
      v16bf bf = load_bfrag(&Bs[(wn*64 + t*16 + lo)*40], hi);
      acc[t] = WMMA_BF16(af, bf, acc[t]);
    }
    __syncthreads();
  }

  const int rbase = mb + wm*16 + 8*hi;
  #pragma unroll
  for (int t = 0; t < 4; t++){
    int col = nb + wn*64 + t*16 + lo;
    if (col >= N) continue;
    #pragma unroll
    for (int r = 0; r < 8; r++){
      int row = rbase + r;
      float v = acc[t][r];
      if (bias) v += bias[col];
      size_t off = (size_t)row*N + col;
      if      (EPI == 0) C[off] = v;
      else if (EPI == 1) C[off] = gelu_tanh(v);
      else {
        int bidx = row / SS;
        C[off] += gate[(size_t)bidx*gstride + col] * v;
      }
    }
  }
}

// ---------------- flash attention (WMMA, online softmax), HD=64, no mask
__global__ __launch_bounds__(256) void attn_kernel(
    const float* __restrict__ qkv, float* __restrict__ o)
{
  __shared__ bf16 Qs[128*72];     // [qrow][dim]
  __shared__ bf16 Ks[32*72];      // [key][dim]
  __shared__ bf16 Vs[64*40];      // [dim][key] (transposed)
  __shared__ bf16 Ps[8*16*40];    // per-wave P scratch [row][key]
  const int tid = threadIdx.x, lane = tid & 31, w = tid >> 5;
  const int hi = lane >> 4, lo = lane & 15;
  const int bh = blockIdx.y, b = bh / HH, h = bh % HH;
  const int q0 = blockIdx.x * 128;
  const size_t rstride = 3*DD;

  for (int idx = tid; idx < 128*64; idx += 256){
    int r = idx >> 6, d = idx & 63;
    float qv = qkv[(size_t)(b*SS + q0 + r)*rstride + h*HDD + d];
    Qs[r*72 + d] = f2bf(qv * ATT_SCALE);
  }
  __syncthreads();
  v16bf qf0 = load_afrag(&Qs[(w*16 + lo)*72], hi);        // dims 0..31
  v16bf qf1 = load_afrag(&Qs[(w*16 + lo)*72 + 32], hi);   // dims 32..63

  v8f o0 = {}, o1 = {}, o2 = {}, o3 = {};
  float mrow[8], lrow[8];
  #pragma unroll
  for (int r = 0; r < 8; r++){ mrow[r] = -1e30f; lrow[r] = 0.f; }

  for (int k0 = 0; k0 < SS; k0 += 32){
    __syncthreads();
    for (int idx = tid; idx < 32*64; idx += 256){
      int r = idx >> 6, d = idx & 63;
      size_t gb = (size_t)(b*SS + k0 + r)*rstride + h*HDD + d;
      Ks[r*72 + d] = f2bf(qkv[gb + DD]);       // K
      Vs[d*40 + r] = f2bf(qkv[gb + 2*DD]);     // V transposed
    }
    __syncthreads();

    v8f s0 = {}, s1 = {};
    s0 = WMMA_BF16(qf0, load_bfrag(&Ks[lo*72],        hi), s0);
    s0 = WMMA_BF16(qf1, load_bfrag(&Ks[lo*72 + 32],   hi), s0);
    s1 = WMMA_BF16(qf0, load_bfrag(&Ks[(16+lo)*72],   hi), s1);
    s1 = WMMA_BF16(qf1, load_bfrag(&Ks[(16+lo)*72+32],hi), s1);

    bf16* pw = &Ps[w*640];
    #pragma unroll
    for (int r = 0; r < 8; r++){
      float smax = fmaxf(s0[r], s1[r]);
      #pragma unroll
      for (int m = 1; m < 16; m <<= 1) smax = fmaxf(smax, __shfl_xor(smax, m, 32));
      float mnew = fmaxf(mrow[r], smax);
      float p0 = __expf(s0[r] - mnew);
      float p1 = __expf(s1[r] - mnew);
      float ps = p0 + p1;
      #pragma unroll
      for (int m = 1; m < 16; m <<= 1) ps += __shfl_xor(ps, m, 32);
      float alpha = __expf(mrow[r] - mnew);
      lrow[r] = lrow[r]*alpha + ps;
      mrow[r] = mnew;
      o0[r] *= alpha; o1[r] *= alpha; o2[r] *= alpha; o3[r] *= alpha;
      int prow = r + 8*hi;
      pw[prow*40 + lo]      = f2bf(p0);
      pw[prow*40 + 16 + lo] = f2bf(p1);
    }
    // per-wave LDS is in-order; re-fragment P into A layout
    v16bf pf = load_afrag(&Ps[w*640 + lo*40], hi);
    o0 = WMMA_BF16(pf, load_bfrag(&Vs[lo*40],      hi), o0);
    o1 = WMMA_BF16(pf, load_bfrag(&Vs[(16+lo)*40], hi), o1);
    o2 = WMMA_BF16(pf, load_bfrag(&Vs[(32+lo)*40], hi), o2);
    o3 = WMMA_BF16(pf, load_bfrag(&Vs[(48+lo)*40], hi), o3);
  }

  #pragma unroll
  for (int r = 0; r < 8; r++){
    int grow = q0 + w*16 + r + 8*hi;
    size_t base = (size_t)(b*SS + grow)*DD + h*HDD;
    float inv = 1.f / lrow[r];
    o[base + 0  + lo] = o0[r]*inv;
    o[base + 16 + lo] = o1[r]*inv;
    o[base + 32 + lo] = o2[r]*inv;
    o[base + 48 + lo] = o3[r]*inv;
  }
}

// ---------------- host orchestration ----------------
extern "C" void kernel_launch(void* const* d_in, const int* in_sizes, int n_in,
                              void* d_out, int out_size, void* d_ws, size_t ws_size,
                              hipStream_t stream) {
  (void)in_sizes; (void)n_in; (void)out_size; (void)ws_size;
  const int*   idx    = (const int*)  d_in[0];
  const float* score  = (const float*)d_in[1];
  const float* sigma  = (const float*)d_in[2];
  const float* embed  = (const float*)d_in[3];
  const float* sw1    = (const float*)d_in[4];
  const float* sb1    = (const float*)d_in[5];
  const float* sw2    = (const float*)d_in[6];
  const float* sb2    = (const float*)d_in[7];
  const float* cw1    = (const float*)d_in[8];
  const float* cb1    = (const float*)d_in[9];
  const float* cw2    = (const float*)d_in[10];
  const float* cb2    = (const float*)d_in[11];
  const float* norm1  = (const float*)d_in[12];
  const float* qkv_w  = (const float*)d_in[13];
  const float* out_w  = (const float*)d_in[14];
  const float* norm2  = (const float*)d_in[15];
  const float* mlp_w1 = (const float*)d_in[16];
  const float* mlp_b1 = (const float*)d_in[17];
  const float* mlp_w2 = (const float*)d_in[18];
  const float* mlp_b2 = (const float*)d_in[19];
  const float* ada_w  = (const float*)d_in[20];
  const float* ada_b  = (const float*)d_in[21];
  const float* normf  = (const float*)d_in[22];
  const float* fin_w  = (const float*)d_in[23];
  const float* fin_b  = (const float*)d_in[24];
  const float* adaf_w = (const float*)d_in[25];
  const float* adaf_b = (const float*)d_in[26];

  // workspace carve (floats)
  float* wc    = (float*)d_ws;            // B*256
  float* wmods = wc    + 1024;            // B*6D (reused for final B*2D)
  float* wx    = wmods + 16384;           // M*D
  float* wh    = wx    + (size_t)MROWS*DD;
  float* wqkv  = wh    + (size_t)MROWS*DD;
  float* wo    = wqkv  + (size_t)MROWS*3*DD;
  float* wmid  = wo    + (size_t)MROWS*DD;

  cond_kernel<<<BB, 256, 0, stream>>>(score, sigma, sw1, sb1, sw2, sb2,
                                      cw1, cb1, cw2, cb2, wc);
  embed_kernel<<<MROWS, 256, 0, stream>>>(idx, embed, wx);

  const dim3 g_qkv((3*DD)/128, MROWS/64), g_d(DD/128, MROWS/64),
             g_mlp(MLPD/128, MROWS/64),  g_fin((VOCABN+127)/128, MROWS/64);

  for (int l = 0; l < LLAYERS; l++){
    const float* n1  = norm1  + (size_t)l*DD;
    const float* n2  = norm2  + (size_t)l*DD;
    const float* qw  = qkv_w  + (size_t)l*DD*3*DD;
    const float* ow  = out_w  + (size_t)l*DD*DD;
    const float* m1w = mlp_w1 + (size_t)l*DD*MLPD;
    const float* m1b = mlp_b1 + (size_t)l*MLPD;
    const float* m2w = mlp_w2 + (size_t)l*MLPD*DD;
    const float* m2b = mlp_b2 + (size_t)l*DD;
    const float* aw  = ada_w  + (size_t)l*CONDN*6*DD;
    const float* ab  = ada_b  + (size_t)l*6*DD;

    ada_kernel<<<dim3((6*DD+255)/256, BB), 256, 0, stream>>>(wc, aw, ab, wmods, 6*DD);
    ln_mod_kernel<<<MROWS, 256, 0, stream>>>(wx, n1, wmods, 0, DD, 6*DD, wh);
    gemm_wmma<0><<<g_qkv, 256, 0, stream>>>(wh, qw, nullptr, nullptr, wqkv, 3*DD, DD, 0);
    rope_kernel<<<MROWS, 256, 0, stream>>>(wqkv);
    attn_kernel<<<dim3(SS/128, BB*HH), 256, 0, stream>>>(wqkv, wo);
    // x += g_m * (o @ out_w)
    gemm_wmma<2><<<g_d, 256, 0, stream>>>(wo, ow, nullptr, wmods + 2*DD, wx, DD, DD, 6*DD);
    ln_mod_kernel<<<MROWS, 256, 0, stream>>>(wx, n2, wmods, 3*DD, 4*DD, 6*DD, wh);
    gemm_wmma<1><<<g_mlp, 256, 0, stream>>>(wh, m1w, m1b, nullptr, wmid, MLPD, DD, 0);
    // x += g_p * (mid @ mlp_w2 + b2)
    gemm_wmma<2><<<g_d, 256, 0, stream>>>(wmid, m2w, m2b, wmods + 5*DD, wx, DD, MLPD, 6*DD);
  }

  // final adaLN + vocab projection
  ada_kernel<<<dim3((2*DD+255)/256, BB), 256, 0, stream>>>(wc, adaf_w, adaf_b, wmods, 2*DD);
  ln_mod_kernel<<<MROWS, 256, 0, stream>>>(wx, normf, wmods, 0, DD, 2*DD, wh);
  gemm_wmma<0><<<g_fin, 256, 0, stream>>>(wh, fin_w, fin_b, nullptr, (float*)d_out,
                                          VOCABN, DD, 0);
}
